// Qwen3Attention_13228499271898
// MI455X (gfx1250) — matrix operand order repaired and checked by reading, not compiled, not run on hardware
//
#include <hip/hip_runtime.h>
#include <hip/hip_bf16.h>
#include <stdint.h>

// ---------------------------------------------------------------------------
// Qwen3 attention block for MI455X (gfx1250, wave32).
// Matmuls: v_wmma_f32_16x16x32_bf16 (8192 MACs/instr, f32 accumulate).
// Data movement: GLOBAL_LOAD_ASYNC_TO_LDS_B128 + s_wait_asynccnt double-buffer.
// ---------------------------------------------------------------------------

#define T_SEQ    2048
#define HIDDEN   4096
#define N_HEADS  32
#define N_KV     8
#define HEAD_DIM 128
#define QKV_N    6144            // 128*(32+16)
#define Q_SIZE   4096            // 32*128
#define KV_SIZE  1024            // 8*128

typedef __bf16   v16bf __attribute__((ext_vector_type(16)));
typedef float    v8f   __attribute__((ext_vector_type(8)));

union FragBF { uint32_t u[8]; v16bf v; };

__device__ __forceinline__ uint16_t f32_to_bf16(float f) {
  union { float f; uint32_t u; } v; v.f = f;
  uint32_t r = v.u + 0x7FFFu + ((v.u >> 16) & 1u);   // round-to-nearest-even
  return (uint16_t)(r >> 16);
}
__device__ __forceinline__ uint32_t pack_bf16x2(float lo, float hi) {
  return (uint32_t)f32_to_bf16(lo) | ((uint32_t)f32_to_bf16(hi) << 16);
}

__device__ __forceinline__ v8f wmma_bf16(v16bf a, v16bf b, v8f c) {
  return __builtin_amdgcn_wmma_f32_16x16x32_bf16(
      false, a, false, b, (short)0, c, false, false);
}

// --- CDNA5 async global->LDS copy (16B per lane), tracked by ASYNCcnt -------
__device__ __forceinline__ uint32_t lds_addr_of(const void* p) {
  return (uint32_t)(uintptr_t)p;   // low 32 bits of LDS aperture addr = offset
}
__device__ __forceinline__ void async_copy_b128(uint32_t lds_addr, const void* gptr) {
  asm volatile("global_load_async_to_lds_b128 %0, %1, off"
               :: "v"(lds_addr), "v"((uint64_t)(uintptr_t)gptr) : "memory");
}
__device__ __forceinline__ void wait_async0() {
  asm volatile("s_wait_asynccnt 0" ::: "memory");
}

// A-fragment (16x32): lane<16: M=lane, K=0..15; lane>=16: M=lane-16, K=16..31.
__device__ __forceinline__ v16bf lds_frag_a(const uint16_t* base, int stride16,
                                            int row_base, int k_off, int lane) {
  const int half = lane >> 4;
  const uint32_t* p = reinterpret_cast<const uint32_t*>(
      base + (size_t)(row_base + (lane & 15)) * stride16 + k_off);
  FragBF f;
  const int b = half * 8;
#pragma unroll
  for (int i = 0; i < 8; ++i) f.u[i] = p[b + i];
  return f.v;
}

// B-fragment (32x16): lane<16: N=lane, K=0..15; lane>=16: N=lane-16, K=16..31.
// LDS holds B transposed, row-major [N][k].
__device__ __forceinline__ v16bf lds_frag_b(const uint16_t* base, int stride16,
                                            int row_base, int k_off, int lane) {
  const int half = lane >> 4;
  const uint32_t* p = reinterpret_cast<const uint32_t*>(
      base + (size_t)(row_base + (lane & 15)) * stride16 + k_off);
  FragBF f;
#pragma unroll
  for (int i = 0; i < 8; ++i) f.u[i] = p[half * 8 + i];
  return f.v;
}

// ---------------------------------------------------------------------------
// Operand prep: f32 -> bf16 (x, and transposed weights so GEMM tiles are
// contiguous-in-K and can be fetched with async b128 copies).
// ---------------------------------------------------------------------------
__global__ __launch_bounds__(256)
void cvt_bf16_kernel(const float* __restrict__ s, uint16_t* __restrict__ d) {
  const size_t i = ((size_t)blockIdx.x * 256 + threadIdx.x) * 2;
  *(uint32_t*)(d + i) = pack_bf16x2(s[i], s[i + 1]);
}

// dst[c][r] = bf16(src[r][c]); 32x32 tiles through LDS.
__global__ __launch_bounds__(256)
void transpose_cvt_kernel(const float* __restrict__ src, uint16_t* __restrict__ dst,
                          int R, int C) {
  __shared__ float tile[32][33];
  const int tx = threadIdx.x & 31, ty = threadIdx.x >> 5;
  const int c0 = blockIdx.x * 32, r0 = blockIdx.y * 32;
#pragma unroll
  for (int i = 0; i < 4; ++i) {
    const int r = ty + i * 8;
    tile[r][tx] = src[(size_t)(r0 + r) * C + c0 + tx];
  }
  __syncthreads();
#pragma unroll
  for (int i = 0; i < 4; ++i) {
    const int r = ty + i * 8;
    dst[(size_t)(c0 + r) * R + r0 + tx] = f32_to_bf16(tile[tx][r]);
  }
}

// ---------------------------------------------------------------------------
// GEMM (TN): C[M,N] f32 = A[M,K]bf16 @ Bt[N,K]bf16ᵀ.
// 128x128 block tile, BK=64, double-buffered async global->LDS staging.
// 8 waves (4Mx2N); wave = 32x64 patch = 2x4 wmma tiles; 16 wmma per barrier.
// ---------------------------------------------------------------------------
#define XSTR 72   // LDS row stride (halfwords): 144B = 9x16B aligned, banks 36l%64 distinct

__global__ __launch_bounds__(256)
void gemm_bf16_tn(const uint16_t* __restrict__ A, const uint16_t* __restrict__ Bt,
                  float* __restrict__ C, int M, int N, int K) {
  __shared__ uint16_t As[2][128 * XSTR];
  __shared__ uint16_t Bs[2][128 * XSTR];

  const int tid  = threadIdx.x;
  const int lane = tid & 31;
  const int wid  = tid >> 5;
  const int wm   = (wid & 3) * 32;
  const int wn   = (wid >> 2) * 64;
  const int m0   = blockIdx.y * 128;
  const int n0   = blockIdx.x * 128;

  v8f acc[2][4];
#pragma unroll
  for (int i = 0; i < 2; ++i)
#pragma unroll
    for (int j = 0; j < 4; ++j) acc[i][j] = (v8f)0.0f;

  // Stage one 128x64 bf16 tile pair: 1024 16B chunks per tile, 4/thread each.
  auto stage = [&](int buf, int kt) {
#pragma unroll
    for (int i = 0; i < 4; ++i) {
      const int ch = tid + i * 256;     // 0..1023
      const int r  = ch >> 3;           // row 0..127
      const int c  = (ch & 7) * 8;      // halfword col {0,8,...,56}
      async_copy_b128(lds_addr_of(&As[buf][r * XSTR + c]),
                      A + (size_t)(m0 + r) * K + kt + c);
      async_copy_b128(lds_addr_of(&Bs[buf][r * XSTR + c]),
                      Bt + (size_t)(n0 + r) * K + kt + c);
    }
  };

  stage(0, 0);
  wait_async0();
  __syncthreads();

  int buf = 0;
  for (int kt = 0; kt < K; kt += 64) {
    if (kt + 64 < K) stage(buf ^ 1, kt + 64);   // overlap fills with wmma
#pragma unroll
    for (int ks = 0; ks < 64; ks += 32) {
      v16bf af[2], bfr[4];
#pragma unroll
      for (int mt = 0; mt < 2; ++mt)
        af[mt] = lds_frag_a(&As[buf][0], XSTR, wm + mt * 16, ks, lane);
#pragma unroll
      for (int nt = 0; nt < 4; ++nt)
        bfr[nt] = lds_frag_b(&Bs[buf][0], XSTR, wn + nt * 16, ks, lane);
#pragma unroll
      for (int mt = 0; mt < 2; ++mt)
#pragma unroll
        for (int nt = 0; nt < 4; ++nt)
          acc[mt][nt] = wmma_bf16(af[mt], bfr[nt], acc[mt][nt]);
    }
    wait_async0();       // next tile resident
    __syncthreads();     // everyone done reading current tile
    buf ^= 1;
  }

  const int half = lane >> 4, nl = lane & 15;
#pragma unroll
  for (int mt = 0; mt < 2; ++mt)
#pragma unroll
    for (int nt = 0; nt < 4; ++nt)
#pragma unroll
      for (int r = 0; r < 8; ++r) {
        const int mrow = m0 + wm + mt * 16 + r + 8 * half;
        const int ncol = n0 + wn + nt * 16 + nl;
        C[(size_t)mrow * N + ncol] = acc[mt][nt][r];
      }
}

// ---------------------------------------------------------------------------
// Per-head RMSNorm + NeoX RoPE, f32 in -> bf16 out. One wave per (t, head).
// ---------------------------------------------------------------------------
__global__ __launch_bounds__(32)
void norm_rope_kernel(const float* __restrict__ qkv, const float* __restrict__ w,
                      const int* __restrict__ pos, uint16_t* __restrict__ out,
                      int nheads, int head_off) {
  const int t = blockIdx.x, h = blockIdx.y, l = threadIdx.x;
  const float* src = qkv + (size_t)t * QKV_N + head_off + h * HEAD_DIM;
  float x0 = src[l], x1 = src[l + 32], x2 = src[l + 64], x3 = src[l + 96];

  float ss = x0 * x0 + x1 * x1 + x2 * x2 + x3 * x3;
#pragma unroll
  for (int off = 1; off < 32; off <<= 1) ss += __shfl_xor(ss, off, 32);
  const float rinv = rsqrtf(ss * (1.0f / 128.0f) + 1e-6f);

  x0 *= rinv * w[l];       x1 *= rinv * w[l + 32];
  x2 *= rinv * w[l + 64];  x3 *= rinv * w[l + 96];

  const float pf = (float)pos[t];
  const float lt = 9.210340371976184f;  // ln(10000)
  const float fr0 = __expf(-((float)(2 * l)        * (1.0f / 128.0f)) * lt);
  const float fr1 = __expf(-((float)(2 * (l + 32)) * (1.0f / 128.0f)) * lt);
  const float a0 = pf * fr0, a1 = pf * fr1;
  const float c0 = __cosf(a0), s0 = __sinf(a0);
  const float c1 = __cosf(a1), s1 = __sinf(a1);

  uint16_t* dst = out + ((size_t)t * nheads + h) * HEAD_DIM;
  dst[l]      = f32_to_bf16(x0 * c0 - x2 * s0);
  dst[l + 64] = f32_to_bf16(x2 * c0 + x0 * s0);
  dst[l + 32] = f32_to_bf16(x1 * c1 - x3 * s1);
  dst[l + 96] = f32_to_bf16(x3 * c1 + x1 * s1);
}

// V slice of qkv -> bf16 [T, N_KV, 128]
__global__ __launch_bounds__(256)
void v_bf16_kernel(const float* __restrict__ qkv, uint16_t* __restrict__ vb) {
  const size_t i = (size_t)blockIdx.x * 256 + threadIdx.x;
  const size_t t = i >> 10, c = i & 1023;
  vb[i] = f32_to_bf16(qkv[t * QKV_N + Q_SIZE + KV_SIZE + c]);
}

// ---------------------------------------------------------------------------
// Flash attention, causal, GQA 4:1. Block = (64 q rows, head), 4 waves; each
// wave owns 16 q rows. Q block and 32-key K chunks staged with async b128
// copies; V transposed via VGPRs. S via 8 WMMAs, online softmax, P reshaped
// through LDS into an A-fragment, PV via 8 WMMAs into 8 v8f accumulators.
// ---------------------------------------------------------------------------
#define KSTR 136   // 272B = 17x16B aligned; banks 68l%64 = 4l distinct
#define PSTR 36

__global__ __launch_bounds__(128)
void attn_kernel(const uint16_t* __restrict__ qb, const uint16_t* __restrict__ kb,
                 const uint16_t* __restrict__ vb, uint16_t* __restrict__ ob) {
  __shared__ uint16_t Qs[64 * KSTR];     // [qrow][d]
  __shared__ uint16_t Ks[32 * KSTR];     // [key][d]
  __shared__ uint16_t Vt[128 * PSTR];    // [d][key]  (transposed for B-frag)
  __shared__ uint16_t Ps[4][16 * PSTR];  // per-wave P tile [m][key]

  const int tid = threadIdx.x, lane = tid & 31, wid = tid >> 5;
  const int h = blockIdx.y, kvh = h >> 2;
  const int q0 = blockIdx.x * 64;
  const int qrow0 = q0 + wid * 16;
  const int half = lane >> 4, nl = lane & 15;
  const float scale = 0.08838834764831845f;  // 1/sqrt(128)

  // Stage Q block 64x128 bf16: 1024 16B chunks / 128 threads.
#pragma unroll
  for (int i = 0; i < 8; ++i) {
    const int ch = tid + i * 128;
    const int r  = ch >> 4;
    const int c  = (ch & 15) * 8;
    async_copy_b128(lds_addr_of(&Qs[r * KSTR + c]),
                    qb + ((size_t)(q0 + r) * N_HEADS + h) * HEAD_DIM + c);
  }
  wait_async0();
  __syncthreads();

  v16bf qf[4];
#pragma unroll
  for (int kk = 0; kk < 4; ++kk)
    qf[kk] = lds_frag_a(Qs, KSTR, wid * 16, kk * 32, lane);

  float m_i[8], l_i[8];
  v8f oacc[8];
#pragma unroll
  for (int r = 0; r < 8; ++r) { m_i[r] = -1e30f; l_i[r] = 0.0f; }
#pragma unroll
  for (int nt = 0; nt < 8; ++nt) oacc[nt] = (v8f)0.0f;

  const int s_end = q0 + 64;
  for (int s0 = 0; s0 < s_end; s0 += 32) {
    __syncthreads();  // all consumers of Ks/Vt from previous chunk are done
    // K chunk 32x128: 512 async chunks / 128 threads.
#pragma unroll
    for (int i = 0; i < 4; ++i) {
      const int ch = tid + i * 128;
      const int r  = ch >> 4;
      const int c  = (ch & 15) * 8;
      async_copy_b128(lds_addr_of(&Ks[r * KSTR + c]),
                      kb + ((size_t)(s0 + r) * N_KV + kvh) * HEAD_DIM + c);
    }
    // V chunk transposed: read [key][d] coalesced, write Vt[d][key].
#pragma unroll
    for (int i = 0; i < 32; ++i) {
      const int e = tid + i * 128;
      const int j = e >> 7, d = e & 127;
      Vt[d * PSTR + j] = vb[((size_t)(s0 + j) * N_KV + kvh) * HEAD_DIM + d];
    }
    wait_async0();
    __syncthreads();

    // S = Q Kᵀ : two 16x16 key tiles, K-dim 128 = 4 wmma steps each.
    v8f c0 = (v8f)0.0f, c1 = (v8f)0.0f;
#pragma unroll
    for (int kk = 0; kk < 4; ++kk) {
      const v16bf b0 = lds_frag_b(Ks, KSTR, 0,  kk * 32, lane);
      const v16bf b1 = lds_frag_b(Ks, KSTR, 16, kk * 32, lane);
      c0 = wmma_bf16(qf[kk], b0, c0);
      c1 = wmma_bf16(qf[kk], b1, c1);
    }

    // Online softmax (row = (half, r); reduce across the 16 N-lanes).
    float p0[8], p1[8], fs[8];
#pragma unroll
    for (int r = 0; r < 8; ++r) {
      const int qr = qrow0 + r + 8 * half;
      const int k0 = s0 + nl, k1 = s0 + 16 + nl;
      const float s0v = (k0 <= qr) ? c0[r] * scale : -1e30f;
      const float s1v = (k1 <= qr) ? c1[r] * scale : -1e30f;
      float mx = fmaxf(s0v, s1v);
#pragma unroll
      for (int off = 1; off < 16; off <<= 1) mx = fmaxf(mx, __shfl_xor(mx, off, 16));
      const float mn = fmaxf(m_i[r], mx);
      const float f  = __expf(m_i[r] - mn);
      const float e0 = (k0 <= qr) ? __expf(s0v - mn) : 0.0f;
      const float e1 = (k1 <= qr) ? __expf(s1v - mn) : 0.0f;
      float rs = e0 + e1;
#pragma unroll
      for (int off = 1; off < 16; off <<= 1) rs += __shfl_xor(rs, off, 16);
      l_i[r] = l_i[r] * f + rs;
      m_i[r] = mn;
      p0[r] = e0; p1[r] = e1; fs[r] = f;
    }
#pragma unroll
    for (int nt = 0; nt < 8; ++nt)
#pragma unroll
      for (int r = 0; r < 8; ++r) oacc[nt][r] *= fs[r];

    // P (C-layout) -> LDS -> A-fragment reshape.
    uint16_t* pp = &Ps[wid][0];
#pragma unroll
    for (int r = 0; r < 8; ++r) {
      const int row = r + 8 * half;
      pp[row * PSTR + nl]      = f32_to_bf16(p0[r]);
      pp[row * PSTR + 16 + nl] = f32_to_bf16(p1[r]);
    }
    asm volatile("s_wait_dscnt 0" ::: "memory");  // same-wave LDS RAW fence

    const v16bf ap = lds_frag_a(pp, PSTR, 0, 0, lane);
#pragma unroll
    for (int nt = 0; nt < 8; ++nt) {
      const v16bf bv = lds_frag_b(Vt, PSTR, nt * 16, 0, lane);
      oacc[nt] = wmma_bf16(ap, bv, oacc[nt]);
    }
  }

  float inv[8];
#pragma unroll
  for (int r = 0; r < 8; ++r) inv[r] = 1.0f / l_i[r];
#pragma unroll
  for (int nt = 0; nt < 8; ++nt)
#pragma unroll
    for (int r = 0; r < 8; ++r) {
      const int qr = qrow0 + r + 8 * half;
      ob[(size_t)qr * HIDDEN + h * HEAD_DIM + nt * 16 + nl] =
          f32_to_bf16(oacc[nt][r] * inv[r]);
    }
}

// ---------------------------------------------------------------------------
extern "C" void kernel_launch(void* const* d_in, const int* in_sizes, int n_in,
                              void* d_out, int out_size, void* d_ws, size_t ws_size,
                              hipStream_t stream) {
  (void)in_sizes; (void)n_in; (void)out_size; (void)ws_size;
  const float* x     = (const float*)d_in[0];
  const int*   pos   = (const int*)d_in[1];
  const float* qkv_w = (const float*)d_in[2];
  const float* qn    = (const float*)d_in[3];
  const float* kn    = (const float*)d_in[4];
  const float* ow    = (const float*)d_in[5];
  float* out = (float*)d_out;

  char* w = (char*)d_ws;
  uint16_t* x_bf    = (uint16_t*)w;  w += (size_t)T_SEQ  * HIDDEN * 2;  // 16.8 MB
  uint16_t* wqkvT   = (uint16_t*)w;  w += (size_t)QKV_N  * HIDDEN * 2;  // 50.3 MB
  uint16_t* owT     = (uint16_t*)w;  w += (size_t)HIDDEN * HIDDEN * 2;  // 33.6 MB
  float*    qkv     = (float*)w;     w += (size_t)T_SEQ * QKV_N * 4;    // 50.3 MB
  uint16_t* q_bf    = (uint16_t*)w;  w += (size_t)T_SEQ * Q_SIZE * 2;   // 16.8 MB
  uint16_t* k_bf    = (uint16_t*)w;  w += (size_t)T_SEQ * KV_SIZE * 2;  //  4.2 MB
  uint16_t* v_bf    = (uint16_t*)w;  w += (size_t)T_SEQ * KV_SIZE * 2;  //  4.2 MB
  uint16_t* attn_bf = (uint16_t*)w;                                     // 16.8 MB

  // 0) operand prep: bf16 activations, transposed bf16 weights
  cvt_bf16_kernel<<<dim3((T_SEQ * HIDDEN / 2) / 256), 256, 0, stream>>>(x, x_bf);
  transpose_cvt_kernel<<<dim3(QKV_N / 32, HIDDEN / 32), 256, 0, stream>>>(
      qkv_w, wqkvT, HIDDEN, QKV_N);
  transpose_cvt_kernel<<<dim3(HIDDEN / 32, HIDDEN / 32), 256, 0, stream>>>(
      ow, owT, HIDDEN, HIDDEN);

  // 1) qkv = x @ qkv_w
  gemm_bf16_tn<<<dim3(QKV_N / 128, T_SEQ / 128), 256, 0, stream>>>(
      x_bf, wqkvT, qkv, T_SEQ, QKV_N, HIDDEN);
  // 2) RMSNorm + RoPE on q and k, emit bf16
  norm_rope_kernel<<<dim3(T_SEQ, N_HEADS), 32, 0, stream>>>(qkv, qn, pos, q_bf, N_HEADS, 0);
  norm_rope_kernel<<<dim3(T_SEQ, N_KV),    32, 0, stream>>>(qkv, kn, pos, k_bf, N_KV, Q_SIZE);
  // 3) v -> bf16
  v_bf16_kernel<<<dim3((T_SEQ * KV_SIZE) / 256), 256, 0, stream>>>(qkv, v_bf);
  // 4) causal flash attention, GQA 4:1
  attn_kernel<<<dim3(T_SEQ / 64, N_HEADS), 128, 0, stream>>>(q_bf, k_bf, v_bf, attn_bf);
  // 5) out = attn @ o_w
  gemm_bf16_tn<<<dim3(HIDDEN / 128, T_SEQ / 128), 256, 0, stream>>>(
      attn_bf, owT, out, T_SEQ, HIDDEN, HIDDEN);
}
